// DartUpdate_46815143526787
// MI455X (gfx1250) — matrix operand order repaired and checked by reading, not compiled, not run on hardware
//
#include <hip/hip_runtime.h>
#include <hip/hip_bf16.h>

typedef __attribute__((ext_vector_type(16))) _Float16 v16h;
typedef __attribute__((ext_vector_type(8)))  _Float16 v8h;
typedef __attribute__((ext_vector_type(8)))  float    v8f;

#define Bn 4
#define Nn 256
#define Hn 128
#define En 32

// packed weight fragment counts (16 halves per lane, 32 lanes per fragment)
#define FR_W1H (32 * 512)   // 16384 halves (32 KB)
#define FR_W1E (8 * 512)    //  4096 halves ( 8 KB)
#define FR_W2  (32 * 512)   // 16384 halves (32 KB)
#define FR_TOT (FR_W1H + FR_W1E + FR_W2)   // 36864 halves = 72 KB
#define HBUF_H (8 * 16 * Hn)               // 16384 halves = 32 KB

// ---------------------------------------------------------------------------
// Kernel 1: S[b,i,k] = b1[k] + sum_h node_h[b,i,h]*W1s[h,k]
//           D[b,j,k] =         sum_h node_h[b,j,h]*W1d[h,k]
// ---------------------------------------------------------------------------
__global__ __launch_bounds__(128) void sd_kernel(const float* __restrict__ node_h,
                                                 const float* __restrict__ W1,
                                                 const float* __restrict__ b1,
                                                 float* __restrict__ S,
                                                 float* __restrict__ D) {
    const int bi = blockIdx.x;      // 0 .. B*N-1
    const int k  = threadIdx.x;     // 0 .. 127
    __shared__ float nh[Hn];
    nh[k] = node_h[(size_t)bi * Hn + k];
    __syncthreads();
    float s = b1[k];
    float d = 0.0f;
#pragma unroll 8
    for (int h = 0; h < Hn; ++h) {
        s = fmaf(nh[h], W1[(size_t)h * Hn + k], s);
        d = fmaf(nh[h], W1[(size_t)(Hn + h) * Hn + k], d);
    }
    S[(size_t)bi * Hn + k] = s;
    D[(size_t)bi * Hn + k] = d;
}

// ---------------------------------------------------------------------------
// Kernel 2: pack W1h (128x128), W1e (32x128), W2 (128x128) into f16
// WMMA B-fragment order (16x16x32 f16): lane -> col n = lane%16,
// element e -> K = ktile*32 + (e&7) + (e>=8 ? 16 : 0) + (lane>=16 ? 8 : 0).
// Linear index within a buffer: ((ktile*8 + ntile)*32 + lane)*16 + e
// The three buffers are CONTIGUOUS: [W1h | W1e | W2]
// ---------------------------------------------------------------------------
__global__ void pack_kernel(const float* __restrict__ W1,
                            const float* __restrict__ W2,
                            _Float16* __restrict__ pw) {
    const int t = blockIdx.x * blockDim.x + threadIdx.x;
    if (t < FR_W1H) {                        // W1h rows 2H..3H-1 of W1
        int e = t & 15, lane = (t >> 4) & 31, frag = t >> 9;
        int kt = frag >> 3, nt = frag & 7;
        int K = kt * 32 + (e & 7) + ((e & 8) ? 16 : 0) + ((lane & 16) ? 8 : 0);
        int n = nt * 16 + (lane & 15);
        pw[t] = (_Float16)W1[(size_t)(2 * Hn + K) * Hn + n];
    } else if (t < FR_W1H + FR_W1E) {        // W1e rows 3H..3H+E-1 of W1
        int t2 = t - FR_W1H;
        int e = t2 & 15, lane = (t2 >> 4) & 31, nt = t2 >> 9;
        int K = (e & 7) + ((e & 8) ? 16 : 0) + ((lane & 16) ? 8 : 0);
        int n = nt * 16 + (lane & 15);
        pw[t] = (_Float16)W1[(size_t)(3 * Hn + K) * Hn + n];
    } else if (t < FR_TOT) {                 // W2
        int t2 = t - (FR_W1H + FR_W1E);
        int e = t2 & 15, lane = (t2 >> 4) & 31, frag = t2 >> 9;
        int kt = frag >> 3, nt = frag & 7;
        int K = kt * 32 + (e & 7) + ((e & 8) ? 16 : 0) + ((lane & 16) ? 8 : 0);
        int n = nt * 16 + (lane & 15);
        pw[t] = (_Float16)W2[(size_t)K * Hn + n];
    }
}

// f16 A-fragment (16x32) from an f32 row pointer already offset to this lane's
// row and K-base. ISA layout: e=0..7 -> K=base+e, e=8..15 -> K=base+16+(e-8).
__device__ inline v16h load_a_f32(const float* __restrict__ p) {
    v16h a;
#pragma unroll
    for (int e = 0; e < 8; ++e) a[e] = (_Float16)p[e];
#pragma unroll
    for (int e = 0; e < 8; ++e) a[8 + e] = (_Float16)p[16 + e];
    return a;
}

// B fragment from LDS-resident packed weights (2x ds_load_b128 per lane)
__device__ inline v16h lds_b_frag(const _Float16* base, int frag, int lane) {
    return *(const v16h*)(base + ((unsigned)frag * 32 + lane) * 16);
}

// ---------------------------------------------------------------------------
// Main fused kernel: one wave per 16-row tile of M = B*N*N (16 consecutive j).
// 8 waves / 256-thread workgroup. Dynamic LDS: 72 KB weights + 32 KB h-stage.
// ---------------------------------------------------------------------------
__global__ __launch_bounds__(256) void dart_main_kernel(
    const float* __restrict__ dart_h, const float* __restrict__ edge_x,
    const float* __restrict__ S, const float* __restrict__ D,
    const _Float16* __restrict__ pw, const float* __restrict__ b2,
    float* __restrict__ out) {

    extern __shared__ _Float16 smem[];
    _Float16* wlds = smem;                 // FR_TOT halves of packed weights
    _Float16* hbuf = smem + FR_TOT;        // HBUF_H halves of h staging

    // ---- stage all packed weights into LDS once per workgroup ----
    {
        const uint4* src = (const uint4*)pw;
        uint4* dst = (uint4*)wlds;
#pragma unroll
        for (int t = 0; t < FR_TOT / 8 / 256; ++t)      // 18 iters, uniform
            dst[t * 256 + threadIdx.x] = src[t * 256 + threadIdx.x];
    }
    __syncthreads();

    const _Float16* w1h = wlds;
    const _Float16* w1e = wlds + FR_W1H;
    const _Float16* w2  = wlds + FR_W1H + FR_W1E;

    const int lane  = threadIdx.x & 31;
    const int wave  = threadIdx.x >> 5;
    const int m0    = (blockIdx.x * 8 + wave) * 16;  // first row of tile
    const int bi    = m0 >> 8;                       // b*N + i (fixed per tile)
    const int b     = bi >> 8;
    const int j0    = m0 & (Nn - 1);
    const int colL  = lane & 15;                     // col (and A-row) in tile
    const int hi    = lane >> 4;
    const int rbase = hi * 8;                        // C/D row offset
    const int koffA = hi * 8;                        // A-frag lane K offset

    _Float16* myh = hbuf + wave * (16 * Hn);

    // ---- A fragments: dart_h (4 ktiles) and edge_x (1 ktile), f32->f16 ----
    v16h aD[4];
    {
        const float* rp = dart_h + (size_t)(m0 + colL) * Hn;
#pragma unroll
        for (int kt = 0; kt < 4; ++kt) aD[kt] = load_a_f32(rp + kt * 32 + koffA);
    }
    v16h aE = load_a_f32(edge_x + (size_t)(m0 + colL) * En + koffA);

    // ---- GEMM1: pre = S + D + dart*W1h + edge*W1e ; h = relu(pre) -> LDS ----
#pragma unroll
    for (int nt = 0; nt < 8; ++nt) {
        const int col = nt * 16 + colL;
        const float sv = S[(size_t)bi * Hn + col];
        v8f acc;
#pragma unroll
        for (int r = 0; r < 8; ++r)
            acc[r] = sv + D[((size_t)(b * Nn) + (j0 + rbase + r)) * Hn + col];

        acc = __builtin_amdgcn_wmma_f32_16x16x32_f16(
                false, aE, false, lds_b_frag(w1e, nt, lane), (short)0, acc, false, false);
#pragma unroll
        for (int kt = 0; kt < 4; ++kt)
            acc = __builtin_amdgcn_wmma_f32_16x16x32_f16(
                    false, aD[kt], false, lds_b_frag(w1h, kt * 8 + nt, lane),
                    (short)0, acc, false, false);

#pragma unroll
        for (int r = 0; r < 8; ++r) {
            float v = acc[r] > 0.0f ? acc[r] : 0.0f;
            myh[(rbase + r) * Hn + col] = (_Float16)v;
        }
    }

    __syncthreads();  // cross-lane C-layout -> A-layout exchange through LDS

    // ---- reload h as A fragments ----
    v16h a2[4];
#pragma unroll
    for (int kt = 0; kt < 4; ++kt) {
        const _Float16* p = myh + colL * Hn + kt * 32 + koffA;
        v8h lo = *(const v8h*)p;
        v8h hh = *(const v8h*)(p + 16);
        v16h a;
#pragma unroll
        for (int e = 0; e < 8; ++e) { a[e] = lo[e]; a[8 + e] = hh[e]; }
        a2[kt] = a;
    }

    // ---- GEMM2: out = h*W2 + b2 ----
#pragma unroll
    for (int nt = 0; nt < 8; ++nt) {
        const int col = nt * 16 + colL;
        const float bb = b2[col];
        v8f acc;
#pragma unroll
        for (int r = 0; r < 8; ++r) acc[r] = bb;
#pragma unroll
        for (int kt = 0; kt < 4; ++kt)
            acc = __builtin_amdgcn_wmma_f32_16x16x32_f16(
                    false, a2[kt], false, lds_b_frag(w2, kt * 8 + nt, lane),
                    (short)0, acc, false, false);
#pragma unroll
        for (int r = 0; r < 8; ++r)
            out[(size_t)(m0 + rbase + r) * Hn + col] = acc[r];
    }
}

// ---------------------------------------------------------------------------
extern "C" void kernel_launch(void* const* d_in, const int* in_sizes, int n_in,
                              void* d_out, int out_size, void* d_ws, size_t ws_size,
                              hipStream_t stream) {
    const float* node_h = (const float*)d_in[0];
    const float* dart_h = (const float*)d_in[1];
    const float* edge_x = (const float*)d_in[2];
    const float* W1     = (const float*)d_in[3];
    const float* b1     = (const float*)d_in[4];
    const float* W2     = (const float*)d_in[5];
    const float* b2     = (const float*)d_in[6];
    float* out = (float*)d_out;

    // workspace carve-up: S (512KB) | D (512KB) | packed weights (72KB)
    float*    S  = (float*)d_ws;
    float*    D  = S + (size_t)Bn * Nn * Hn;
    _Float16* pw = (_Float16*)(D + (size_t)Bn * Nn * Hn);

    const int dynLds = (FR_TOT + HBUF_H) * (int)sizeof(_Float16);  // 104 KB
    (void)hipFuncSetAttribute((const void*)dart_main_kernel,
                              hipFuncAttributeMaxDynamicSharedMemorySize, dynLds);

    sd_kernel<<<Bn * Nn, Hn, 0, stream>>>(node_h, W1, b1, S, D);
    pack_kernel<<<(FR_TOT + 255) / 256, 256, 0, stream>>>(W1, W2, pw);

    const int tiles = (Bn * Nn * Nn) / 16;           // 16384 wave-tiles
    dart_main_kernel<<<tiles / 8, 256, dynLds, stream>>>(dart_h, edge_x, S, D,
                                                         pw, b2, out);
}
